// CL4SRecEncoder_56667798503735
// MI455X (gfx1250) — matrix operand order repaired and checked by reading, not compiled, not run on hardware
//
#include <hip/hip_runtime.h>
#include <hip/hip_bf16.h>
#include <math.h>

#define B_  128
#define L_  256
#define D_  256
#define H_  8
#define DK_ 32
#define DFF_ 1024
#define NBLK_ 2
#define NUM_ITEMS_ 50000
#define M_  (B_ * L_)   // 32768 tokens

typedef __attribute__((ext_vector_type(16))) __bf16 v16bf;
typedef __attribute__((ext_vector_type(8)))  __bf16 v8bf;
typedef __attribute__((ext_vector_type(8)))  float  v8f;
typedef __attribute__((ext_vector_type(4)))  int    v4i;
typedef __bf16 bf16_t;

#if defined(__HIP_DEVICE_COMPILE__) && __has_builtin(__builtin_amdgcn_global_load_async_to_lds_b128)
#define HAVE_ASYNC_LDS 1
#else
#define HAVE_ASYNC_LDS 0
#endif

__device__ inline v16bf cat8(v8bf lo, v8bf hi) {
  return __builtin_shufflevector(lo, hi, 0,1,2,3,4,5,6,7,8,9,10,11,12,13,14,15);
}

__device__ inline v8f wmma_bf16(v16bf a, v16bf b, v8f c) {
  // D = A(16x32 bf16) * B(32x16 bf16) + C(16x16 f32)
  return __builtin_amdgcn_wmma_f32_16x16x32_bf16(false, a, false, b, (short)0, c, false, false);
}

// ---------------------------------------------------------------------------
// Embedding gather: x[tok] = item_emb[clip(seq[tok]+1)] + pos_emb[tok % L]
// ---------------------------------------------------------------------------
__global__ __launch_bounds__(256) void embed_kernel(
    const int* __restrict__ seq, const float* __restrict__ item_emb,
    const float* __restrict__ pos_emb, float* __restrict__ x) {
  int idx = blockIdx.x * blockDim.x + threadIdx.x;   // over M_*64
  int tok = idx >> 6;
  int c4  = idx & 63;
  int l   = tok & (L_ - 1);
  int id  = seq[tok] + 1;
  if (id < 0) id = 0;
  if (id > NUM_ITEMS_) id = NUM_ITEMS_;
  float4 e = ((const float4*)item_emb)[(size_t)id * 64 + c4];
  float4 p = ((const float4*)pos_emb)[(size_t)l * 64 + c4];
  float4 r;
  r.x = e.x + p.x; r.y = e.y + p.y; r.z = e.z + p.z; r.w = e.w + p.w;
  ((float4*)x)[idx] = r;
}

// ---------------------------------------------------------------------------
// Weight prep: f32 [K][N] row-major -> bf16 [N][K] row-major (transposed)
// ---------------------------------------------------------------------------
__global__ __launch_bounds__(256) void transpose_bf16_kernel(
    const float* __restrict__ in, bf16_t* __restrict__ out, int K, int N) {
  int idx = blockIdx.x * blockDim.x + threadIdx.x;
  if (idx >= K * N) return;
  int k = idx / N, n = idx % N;
  out[(size_t)n * K + k] = (bf16_t)in[idx];
}

// ---------------------------------------------------------------------------
// LayerNorm over D=256, one wave per row (8 floats per lane).
// ---------------------------------------------------------------------------
__global__ __launch_bounds__(256) void ln_kernel(
    const float* __restrict__ x, const float* __restrict__ g,
    const float* __restrict__ b, bf16_t* __restrict__ outB,
    float* __restrict__ outF) {
  int wid = threadIdx.x >> 5, lane = threadIdx.x & 31;
  int row = blockIdx.x * 8 + wid;
  const float* xr = x + (size_t)row * D_;
  float v[8];
  float4 a0 = ((const float4*)xr)[lane * 2];
  float4 a1 = ((const float4*)xr)[lane * 2 + 1];
  v[0]=a0.x; v[1]=a0.y; v[2]=a0.z; v[3]=a0.w;
  v[4]=a1.x; v[5]=a1.y; v[6]=a1.z; v[7]=a1.w;
  float s = 0.f;
#pragma unroll
  for (int i = 0; i < 8; i++) s += v[i];
#pragma unroll
  for (int m = 1; m < 32; m <<= 1) s += __shfl_xor(s, m, 32);
  float mu = s * (1.0f / 256.0f);
  float vs = 0.f;
#pragma unroll
  for (int i = 0; i < 8; i++) { float d = v[i] - mu; vs += d * d; }
#pragma unroll
  for (int m = 1; m < 32; m <<= 1) vs += __shfl_xor(vs, m, 32);
  float rstd = rsqrtf(vs * (1.0f / 256.0f) + 1e-5f);
  int cbase = lane * 8;
#pragma unroll
  for (int i = 0; i < 8; i++) {
    float y = (v[i] - mu) * rstd * g[cbase + i] + b[cbase + i];
    size_t idx = (size_t)row * D_ + cbase + i;
    if (outB) outB[idx] = (bf16_t)y;
    if (outF) outF[idx] = y;
  }
}

// ---------------------------------------------------------------------------
// WMMA GEMM: C[M,N] = A[M,K](bf16,row-major) * Wt[N,K](bf16,N-major) + bias
// Block: 256 threads = 8 waves; block tile = 256(M) x 32(N).
// Wave tile = 32x32 (2x2 WMMA). B tile (32 x Kchunk) staged in LDS via
// async-to-LDS loads (ASYNCcnt) with padded pitch to avoid bank conflicts.
// Epilogue: optional exact GELU, optional f32 residual add, f32/bf16 out.
// ---------------------------------------------------------------------------
__global__ __launch_bounds__(256) void gemm_wmma_kernel(
    const bf16_t* __restrict__ A, const bf16_t* __restrict__ Wt,
    const float* __restrict__ bias, const float* __restrict__ resid,
    float* __restrict__ outF, bf16_t* __restrict__ outB,
    int M, int N, int K, int apply_gelu) {
  extern __shared__ __align__(16) bf16_t ldsB[];   // 32 * (KC+8) bf16

  const int tid  = threadIdx.x;
  const int wid  = tid >> 5;
  const int lane = tid & 31;
  const int col  = lane & 15;        // A-frag row / B-frag col / C col
  const int half = lane >> 4;
  const int n0 = blockIdx.x * 32;
  const int m0 = (blockIdx.y * 8 + wid) * 32;

  const int KC    = (K > 512) ? 512 : K;   // K staging chunk
  const int pitch = KC + 8;                // bf16 elems; %64-dword offset = 4

  v8f c00, c01, c10, c11;
  const float bv0 = bias[n0 + col];
  const float bv1 = bias[n0 + 16 + col];
#pragma unroll
  for (int i = 0; i < 8; i++) { c00[i] = bv0; c10[i] = bv0; c01[i] = bv1; c11[i] = bv1; }

  const bf16_t* a0Base = A + (size_t)(m0 + col) * K + half * 8;
  const bf16_t* a1Base = a0Base + (size_t)16 * K;

  for (int kc0 = 0; kc0 < K; kc0 += KC) {
    // ---- stage B tile: rows n0..n0+31 of Wt, KC elems each (contiguous rows)
    const int chunksPerRow = KC / 8;                 // 16-byte chunks
    for (int cidx = tid; cidx < 32 * chunksPerRow; cidx += 256) {
      int r = cidx / chunksPerRow, w8 = cidx % chunksPerRow;
      const bf16_t* src = Wt + (size_t)(n0 + r) * K + kc0 + w8 * 8;
      bf16_t* dst = ldsB + (size_t)r * pitch + w8 * 8;
#if HAVE_ASYNC_LDS
      __builtin_amdgcn_global_load_async_to_lds_b128(
          (__attribute__((address_space(1))) v4i*)const_cast<bf16_t*>(src),
          (__attribute__((address_space(3))) v4i*)dst, 0, 0);
#else
      *(v8bf*)dst = *(const v8bf*)src;
#endif
    }
#if HAVE_ASYNC_LDS
#if __has_builtin(__builtin_amdgcn_s_wait_asynccnt)
    __builtin_amdgcn_s_wait_asynccnt(0);
#else
    asm volatile("s_wait_asynccnt 0x0" ::: "memory");
#endif
#endif
    __syncthreads();

    // ---- inner product over this K chunk
    for (int k = 0; k < KC; k += 32) {
      __builtin_prefetch(a0Base + kc0 + k + 128, 0, 0);
      __builtin_prefetch(a1Base + kc0 + k + 128, 0, 0);
      v16bf a0 = cat8(*(const v8bf*)(a0Base + kc0 + k),
                      *(const v8bf*)(a0Base + kc0 + k + 16));
      v16bf a1 = cat8(*(const v8bf*)(a1Base + kc0 + k),
                      *(const v8bf*)(a1Base + kc0 + k + 16));
      const bf16_t* b0p = ldsB + (size_t)col * pitch + half * 8 + k;
      const bf16_t* b1p = b0p + (size_t)16 * pitch;
      v16bf b0 = cat8(*(const v8bf*)b0p, *(const v8bf*)(b0p + 16));
      v16bf b1 = cat8(*(const v8bf*)b1p, *(const v8bf*)(b1p + 16));
      c00 = wmma_bf16(a0, b0, c00);
      c01 = wmma_bf16(a0, b1, c01);
      c10 = wmma_bf16(a1, b0, c10);
      c11 = wmma_bf16(a1, b1, c11);
    }
    __syncthreads();   // protect LDS before next staging round
  }

  // ---- epilogue: 4 tiles
  v8f acc[2][2] = {{c00, c01}, {c10, c11}};
#pragma unroll
  for (int mt = 0; mt < 2; mt++) {
#pragma unroll
    for (int nt = 0; nt < 2; nt++) {
      const int rbase = m0 + mt * 16 + half * 8;
      const int ncol  = n0 + nt * 16 + col;
#pragma unroll
      for (int v = 0; v < 8; v++) {
        float val = acc[mt][nt][v];
        if (apply_gelu) val = 0.5f * val * (1.0f + erff(val * 0.70710678118654752f));
        size_t idx = (size_t)(rbase + v) * N + ncol;
        if (resid) val += resid[idx];
        if (outF) outF[idx] = val;
        if (outB) outB[idx] = (bf16_t)val;
      }
    }
  }
}

// ---------------------------------------------------------------------------
// Flash-style causal attention, one wave per 16-query tile per (b,h).
// Q/K/V bf16 laid out [B*L][D] with head h at columns h*32 .. h*32+31.
// Scores + P@V via WMMA; P transposed C-layout -> A-layout through LDS.
// ---------------------------------------------------------------------------
__global__ __launch_bounds__(256) void attn_kernel(
    const bf16_t* __restrict__ Qb, const bf16_t* __restrict__ Kb,
    const bf16_t* __restrict__ Vb, const int* __restrict__ mask,
    bf16_t* __restrict__ Ob) {
  __shared__ __align__(16) bf16_t lds_p[8][16 * 32];   // per-wave P tile

  const int wid  = threadIdx.x >> 5;
  const int lane = threadIdx.x & 31;
  const int col  = lane & 15;
  const int half = lane >> 4;
  const int bidx = blockIdx.z;
  const int hidx = blockIdx.y;
  const int q0   = (blockIdx.x * 8 + wid) * 16;

  const size_t base = (size_t)bidx * L_ * D_;
  const int hoff = hidx * DK_;
  const float scale = 0.17677669529663689f;   // 1/sqrt(32)

  // Q fragment (A-layout): row = lane&15, kbase = half*8
  const bf16_t* qptr = Qb + base + (size_t)(q0 + col) * D_ + hoff + half * 8;
  v16bf qf = cat8(*(const v8bf*)qptr, *(const v8bf*)(qptr + 16));

  float mrun[8], lrun[8];
  v8f o0 = {0,0,0,0,0,0,0,0}, o1 = {0,0,0,0,0,0,0,0};
#pragma unroll
  for (int v = 0; v < 8; v++) { mrun[v] = -1e30f; lrun[v] = 0.f; }

  for (int c0 = 0; c0 < q0 + 16; c0 += 32) {
    // K^T B-fragments for two 16-key tiles: col = key, dk base half*8, second 8 at +16
    const bf16_t* kp0 = Kb + base + (size_t)(c0 + col) * D_ + hoff + half * 8;
    const bf16_t* kp1 = kp0 + (size_t)16 * D_;
    v16bf kf0 = cat8(*(const v8bf*)kp0, *(const v8bf*)(kp0 + 16));
    v16bf kf1 = cat8(*(const v8bf*)kp1, *(const v8bf*)(kp1 + 16));

    v8f s0 = {0,0,0,0,0,0,0,0}, s1 = {0,0,0,0,0,0,0,0};
    s0 = wmma_bf16(qf, kf0, s0);
    s1 = wmma_bf16(qf, kf1, s1);

    const int key0 = c0 + col, key1 = key0 + 16;
    const bool pad0 = (mask[bidx * L_ + key0] == 0);
    const bool pad1 = (mask[bidx * L_ + key1] == 0);

    float p0[8], p1[8], alpha[8];
#pragma unroll
    for (int v = 0; v < 8; v++) {
      int qg = q0 + half * 8 + v;
      float a = s0[v] * scale; if (key0 > qg || pad0) a = -1e9f;
      float e = s1[v] * scale; if (key1 > qg || pad1) e = -1e9f;
      float rm = fmaxf(a, e);
#pragma unroll
      for (int m = 1; m < 16; m <<= 1) rm = fmaxf(rm, __shfl_xor(rm, m, 32));
      float mnew = fmaxf(mrun[v], rm);
      float al = __expf(mrun[v] - mnew);
      float e0 = __expf(a - mnew);
      float e1 = __expf(e - mnew);
      float rs = e0 + e1;
#pragma unroll
      for (int m = 1; m < 16; m <<= 1) rs += __shfl_xor(rs, m, 32);
      lrun[v] = lrun[v] * al + rs;
      mrun[v] = mnew;
      alpha[v] = al; p0[v] = e0; p1[v] = e1;
    }

    // rescale O, stash P into LDS in [row][relkey] order
    bf16_t* pl = lds_p[wid];
#pragma unroll
    for (int v = 0; v < 8; v++) {
      o0[v] *= alpha[v];
      o1[v] *= alpha[v];
      int r = half * 8 + v;
      pl[r * 32 + col]      = (bf16_t)p0[v];
      pl[r * 32 + col + 16] = (bf16_t)p1[v];
    }

    // reload P as A-fragment (transpose across lanes via LDS)
    const bf16_t* pf = pl + col * 32 + half * 8;
    v16bf pfrag = cat8(*(const v8bf*)pf, *(const v8bf*)(pf + 16));

    // V B-fragments (col = dk, keys strided by D_ in memory)
    const bf16_t* vbase = Vb + base + hoff;
    const int koffk = half * 8;
    v16bf vf0, vf1;
#pragma unroll
    for (int i = 0; i < 16; i++) {
      size_t roff = (size_t)(c0 + koffk + (i < 8 ? i : i + 8)) * D_;
      vf0[i] = vbase[roff + col];
      vf1[i] = vbase[roff + 16 + col];
    }
    o0 = wmma_bf16(pfrag, vf0, o0);
    o1 = wmma_bf16(pfrag, vf1, o1);
  }

#pragma unroll
  for (int v = 0; v < 8; v++) {
    float inv = 1.0f / lrun[v];
    size_t r = base + (size_t)(q0 + half * 8 + v) * D_ + hoff;
    Ob[r + col]      = (bf16_t)(o0[v] * inv);
    Ob[r + 16 + col] = (bf16_t)(o1[v] * inv);
  }
}

// ---------------------------------------------------------------------------
extern "C" void kernel_launch(void* const* d_in, const int* in_sizes, int n_in,
                              void* d_out, int out_size, void* d_ws, size_t ws_size,
                              hipStream_t stream) {
  (void)in_sizes; (void)n_in; (void)out_size; (void)ws_size;
  const int*   seq      = (const int*)  d_in[0];
  const int*   mask     = (const int*)  d_in[1];
  const float* item_emb = (const float*)d_in[2];
  const float* pos_emb  = (const float*)d_in[3];
  const float* Wq = (const float*)d_in[4];  const float* bq = (const float*)d_in[5];
  const float* Wk = (const float*)d_in[6];  const float* bk = (const float*)d_in[7];
  const float* Wv = (const float*)d_in[8];  const float* bv = (const float*)d_in[9];
  const float* Wo = (const float*)d_in[10]; const float* bo = (const float*)d_in[11];
  const float* ln1_g = (const float*)d_in[12]; const float* ln1_b = (const float*)d_in[13];
  const float* W1 = (const float*)d_in[14]; const float* b1 = (const float*)d_in[15];
  const float* W2 = (const float*)d_in[16]; const float* b2 = (const float*)d_in[17];
  const float* ln2_g = (const float*)d_in[18]; const float* ln2_b = (const float*)d_in[19];
  const float* fn_g  = (const float*)d_in[20]; const float* fn_b  = (const float*)d_in[21];

  // Workspace layout
  char* ws = (char*)d_ws;
  float*  x   = (float*)ws;                                   // 32768*256 f32 = 33.5 MB
  bf16_t* xn  = (bf16_t*)(ws + 33554432);                     // 16.8 MB
  bf16_t* wts = (bf16_t*)(ws + 33554432 + 16777216);          // 3.1 MB bf16 weights
  char* scratch = ws + 33554432 + 16777216 + 3145728;         // 67.1 MB
  bf16_t* Qb = (bf16_t*)scratch;
  bf16_t* Kb = (bf16_t*)(scratch + 16777216);
  bf16_t* Vb = (bf16_t*)(scratch + 33554432);
  bf16_t* Ob = (bf16_t*)(scratch + 50331648);
  bf16_t* Hb = (bf16_t*)scratch;            // FFN hidden reuses Q/K/V/O region

  // 1) embedding gather
  embed_kernel<<<(M_ * 64) / 256, 256, 0, stream>>>(seq, item_emb, pos_emb, x);

  // 2) weight prep: transpose+convert to bf16 [N][K]
  for (int i = 0; i < NBLK_; i++) {
    size_t wb = (size_t)i * 786432;
    transpose_bf16_kernel<<<256, 256, 0, stream>>>(Wq + (size_t)i*65536,  wts + wb,          D_, D_);
    transpose_bf16_kernel<<<256, 256, 0, stream>>>(Wk + (size_t)i*65536,  wts + wb + 65536,  D_, D_);
    transpose_bf16_kernel<<<256, 256, 0, stream>>>(Wv + (size_t)i*65536,  wts + wb + 131072, D_, D_);
    transpose_bf16_kernel<<<256, 256, 0, stream>>>(Wo + (size_t)i*65536,  wts + wb + 196608, D_, D_);
    transpose_bf16_kernel<<<1024, 256, 0, stream>>>(W1 + (size_t)i*262144, wts + wb + 262144, D_, DFF_);
    transpose_bf16_kernel<<<1024, 256, 0, stream>>>(W2 + (size_t)i*262144, wts + wb + 524288, DFF_, D_);
  }

  const dim3 ggD(D_ / 32, M_ / 256);      // N=256 GEMMs
  const dim3 ggF(DFF_ / 32, M_ / 256);    // N=1024 GEMM
  const size_t smemK256  = 32 * (256 + 8) * sizeof(bf16_t);   // K=256 chunk
  const size_t smemK1024 = 32 * (512 + 8) * sizeof(bf16_t);   // K=1024, chunked at 512

  for (int i = 0; i < NBLK_; i++) {
    size_t wb = (size_t)i * 786432;
    // attention sublayer
    ln_kernel<<<M_ / 8, 256, 0, stream>>>(x, ln1_g + i*D_, ln1_b + i*D_, xn, nullptr);
    gemm_wmma_kernel<<<ggD, 256, smemK256, stream>>>(xn, wts + wb,          bq + i*D_, nullptr, nullptr, Qb, M_, D_, D_, 0);
    gemm_wmma_kernel<<<ggD, 256, smemK256, stream>>>(xn, wts + wb + 65536,  bk + i*D_, nullptr, nullptr, Kb, M_, D_, D_, 0);
    gemm_wmma_kernel<<<ggD, 256, smemK256, stream>>>(xn, wts + wb + 131072, bv + i*D_, nullptr, nullptr, Vb, M_, D_, D_, 0);
    attn_kernel<<<dim3(L_ / 128, H_, B_), 256, 0, stream>>>(Qb, Kb, Vb, mask, Ob);
    gemm_wmma_kernel<<<ggD, 256, smemK256, stream>>>(Ob, wts + wb + 196608, bo + i*D_, x, x, nullptr, M_, D_, D_, 0);
    // FFN sublayer
    ln_kernel<<<M_ / 8, 256, 0, stream>>>(x, ln2_g + i*D_, ln2_b + i*D_, xn, nullptr);
    gemm_wmma_kernel<<<ggF, 256, smemK256, stream>>>(xn, wts + wb + 262144, b1 + i*DFF_, nullptr, nullptr, Hb, M_, DFF_, D_, 1);
    gemm_wmma_kernel<<<ggD, 256, smemK1024, stream>>>(Hb, wts + wb + 524288, b2 + i*D_, x, x, nullptr, M_, D_, DFF_, 0);
  }

  // 3) final layernorm -> f32 output
  ln_kernel<<<M_ / 8, 256, 0, stream>>>(x, fn_g, fn_b, nullptr, (float*)d_out);
}
